// GCN_icml_2019_20727512170682
// MI455X (gfx1250) — matrix operand order repaired and checked by reading, not compiled, not run on hardware
//
#include <hip/hip_runtime.h>
#include <hip/hip_bf16.h>

// ---------------------------------------------------------------------------
// GCN (ICML'19 style) forward: 2-hop SpMM propagation + WMMA MLP filter +
// rating loss. MI455X reasoning: SpMM traffic (~2GB irregular) is HBM-bound;
// the dense MLP (6.6 GFLOP, AI~64 FLOP/B when fused) runs on the matrix cores
// via v_wmma_f32_16x16x32_f16 with fp32 accumulation.
// ---------------------------------------------------------------------------

#define U_CNT 200000
#define I_CNT 100000
#define DD    64
#define HH    128
#define NNZ_CNT 1000000
#define B_CNT 8192
#define LAMADA 0.001f

typedef _Float16 v16h __attribute__((ext_vector_type(16)));
typedef float    v8f  __attribute__((ext_vector_type(8)));

// ---------------------------------------------------------------------------
__global__ void zero_kernel(float* __restrict__ p, size_t n) {
    size_t i = (size_t)blockIdx.x * blockDim.x + threadIdx.x;
    if (i < n) p[i] = 0.0f;
}

// out[dst[e], :] += vals[e] * x[src[e], :]   (one wave per edge, float2/lane)
__global__ void spmm_kernel(float* __restrict__ out, const float* __restrict__ x,
                            const int* __restrict__ src, const int* __restrict__ dst,
                            const float* __restrict__ vals, int nnz) {
    int gid  = blockIdx.x * blockDim.x + threadIdx.x;
    int e    = gid >> 5;
    int lane = gid & 31;
    if (e >= nnz) return;
    int s = src[e];
    int d = dst[e];
    float v = vals[e];
    float2 xx = reinterpret_cast<const float2*>(x + (size_t)s * DD)[lane];
    atomicAdd(out + (size_t)d * DD + 2 * lane,     v * xx.x);
    atomicAdd(out + (size_t)d * DD + 2 * lane + 1, v * xx.y);
}

// g1 = relu(spmm + x * deg[row])
__global__ void ew_hop1_kernel(float* __restrict__ out, const float* __restrict__ s,
                               const float* __restrict__ x, const float* __restrict__ deg,
                               size_t n) {
    size_t i = (size_t)blockIdx.x * blockDim.x + threadIdx.x;
    if (i >= n) return;
    float v = s[i] + x[i] * deg[i >> 6];
    out[i] = v > 0.0f ? v : 0.0f;
}

// g = base + g1 + relu(spmm + g1 * deg[row])   (fused hop2 + 3-term sum)
__global__ void ew_hop2_kernel(float* __restrict__ out, const float* __restrict__ base,
                               const float* __restrict__ g1, const float* __restrict__ s,
                               const float* __restrict__ deg, size_t n) {
    size_t i = (size_t)blockIdx.x * blockDim.x + threadIdx.x;
    if (i >= n) return;
    float g = g1[i];
    float v = s[i] + g * deg[i >> 6];
    v = v > 0.0f ? v : 0.0f;
    out[i] = base[i] + g + v;
}

// ---------------------------------------------------------------------------
// Fused MLP: guf = leaky(leaky(gu@W1 + b1)@W2 + b2), leaky slope 0.1.
// 128 threads = 4 waves; each wave owns a 16-row tile; weights live in LDS
// transposed ([n][k]) and converted to f16 so fragment K-pairs are contiguous.
// Fragment layouts follow CDNA5 ISA §7.12.2 (wave32).
// ---------------------------------------------------------------------------
__launch_bounds__(128)
__global__ void mlp_wmma_kernel(const float* __restrict__ gu,
                                const float* __restrict__ W1, const float* __restrict__ b1,
                                const float* __restrict__ W2, const float* __restrict__ b2,
                                float* __restrict__ guf, int nRows) {
    __shared__ _Float16 w1t[HH * DD];        // W1^T : [n<128][k<64]
    __shared__ _Float16 w2t[DD * HH];        // W2^T : [n<64][k<128]
    __shared__ float    b1s[HH];
    __shared__ float    b2s[DD];
    __shared__ _Float16 aStage[4][16 * DD];  // per-wave input tile  16x64
    __shared__ _Float16 hStage[4][16 * HH];  // per-wave hidden tile 16x128

    const int tid  = threadIdx.x;
    const int wave = tid >> 5;
    const int lane = tid & 31;
    const int nl   = lane & 15;   // column / M index within 16
    const int hiH  = lane >> 4;   // half-wave selector

    for (int idx = tid; idx < HH * DD; idx += blockDim.x) {
        int n = idx >> 6, k = idx & 63;
        w1t[idx] = (_Float16)W1[k * HH + n];
    }
    for (int idx = tid; idx < DD * HH; idx += blockDim.x) {
        int n = idx >> 7, k = idx & 127;
        w2t[idx] = (_Float16)W2[k * DD + n];
    }
    for (int idx = tid; idx < HH; idx += blockDim.x) b1s[idx] = b1[idx];
    for (int idx = tid; idx < DD; idx += blockDim.x) b2s[idx] = b2[idx];

    // stage this wave's 16x64 fp32 rows -> f16 LDS tile (coalesced)
    const int row0 = blockIdx.x * 64 + wave * 16;
    for (int i = 0; i < 32; ++i) {
        int flat = lane + i * 32;
        int rr = flat >> 6, cc = flat & 63;
        int row = row0 + rr;
        aStage[wave][flat] = (row < nRows) ? (_Float16)gu[(size_t)row * DD + cc]
                                           : (_Float16)0.0f;
    }
    __syncthreads();

    // A fragments for GEMM1 (two K=32 chunks of the 16x64 tile)
    v16h aF[2];
    #pragma unroll
    for (int kc = 0; kc < 2; ++kc) {
        #pragma unroll
        for (int v = 0; v < 8; ++v) {
            int kb = ((v < 4) ? (2 * v + 8 * hiH) : (16 + 2 * (v - 4) + 8 * hiH)) + 32 * kc;
            aF[kc][2 * v]     = aStage[wave][nl * DD + kb];
            aF[kc][2 * v + 1] = aStage[wave][nl * DD + kb + 1];
        }
    }

    // GEMM1: h[16x128] = A @ W1, bias + leaky, into hStage
    #pragma unroll
    for (int nt = 0; nt < 8; ++nt) {
        const int nBase = nt * 16;
        v8f acc = {};
        #pragma unroll
        for (int kc = 0; kc < 2; ++kc) {
            v16h bF;
            #pragma unroll
            for (int v = 0; v < 8; ++v) {
                int kb = ((v < 4) ? (2 * v + 8 * hiH) : (16 + 2 * (v - 4) + 8 * hiH)) + 32 * kc;
                bF[2 * v]     = w1t[(nBase + nl) * DD + kb];
                bF[2 * v + 1] = w1t[(nBase + nl) * DD + kb + 1];
            }
            acc = __builtin_amdgcn_wmma_f32_16x16x32_f16(false, aF[kc], false, bF,
                                                         (short)0, acc, false, false);
        }
        const float bias = b1s[nBase + nl];
        #pragma unroll
        for (int r = 0; r < 8; ++r) {
            float val = acc[r] + bias;
            val = (val > 0.0f) ? val : 0.1f * val;
            hStage[wave][(r + 8 * hiH) * HH + nBase + nl] = (_Float16)val;
        }
    }
    __syncthreads();

    // A fragments for GEMM2 (four K=32 chunks of the 16x128 hidden tile)
    v16h aF2[4];
    #pragma unroll
    for (int kc = 0; kc < 4; ++kc) {
        #pragma unroll
        for (int v = 0; v < 8; ++v) {
            int kb = ((v < 4) ? (2 * v + 8 * hiH) : (16 + 2 * (v - 4) + 8 * hiH)) + 32 * kc;
            aF2[kc][2 * v]     = hStage[wave][nl * HH + kb];
            aF2[kc][2 * v + 1] = hStage[wave][nl * HH + kb + 1];
        }
    }

    // GEMM2: out[16x64] = h @ W2, bias + leaky, store to guf
    #pragma unroll
    for (int nt = 0; nt < 4; ++nt) {
        const int nBase = nt * 16;
        v8f acc = {};
        #pragma unroll
        for (int kc = 0; kc < 4; ++kc) {
            v16h bF;
            #pragma unroll
            for (int v = 0; v < 8; ++v) {
                int kb = ((v < 4) ? (2 * v + 8 * hiH) : (16 + 2 * (v - 4) + 8 * hiH)) + 32 * kc;
                bF[2 * v]     = w2t[(nBase + nl) * HH + kb];
                bF[2 * v + 1] = w2t[(nBase + nl) * HH + kb + 1];
            }
            acc = __builtin_amdgcn_wmma_f32_16x16x32_f16(false, aF2[kc], false, bF,
                                                         (short)0, acc, false, false);
        }
        const float bias = b2s[nBase + nl];
        #pragma unroll
        for (int r = 0; r < 8; ++r) {
            int row = row0 + r + 8 * hiH;
            if (row < nRows) {
                float val = acc[r] + bias;
                val = (val > 0.0f) ? val : 0.1f * val;
                guf[(size_t)row * DD + nBase + nl] = val;
            }
        }
    }
}

// ---------------------------------------------------------------------------
// One wave per sample: 64-dim dot + L2 partials, shuffle-reduce, atomics.
__global__ void predict_kernel(const float* __restrict__ guf, const float* __restrict__ gi,
                               const int* __restrict__ user0, const int* __restrict__ item0,
                               const float* __restrict__ ubias, const float* __restrict__ ibias,
                               const float* __restrict__ avg, const float* __restrict__ ratings,
                               float* __restrict__ acc, int nB) {
    int gid  = blockIdx.x * blockDim.x + threadIdx.x;
    int b    = gid >> 5;
    int lane = gid & 31;
    if (b >= nB) return;
    int u  = user0[b];
    int it = item0[b];
    const float* up = guf + (size_t)u  * DD;
    const float* ip = gi  + (size_t)it * DD;
    float u0 = up[lane], u1 = up[lane + 32];
    float i0 = ip[lane], i1 = ip[lane + 32];
    float dot = u0 * i0 + u1 * i1;
    float us  = u0 * u0 + u1 * u1;
    float is  = i0 * i0 + i1 * i1;
    #pragma unroll
    for (int off = 16; off > 0; off >>= 1) {
        dot += __shfl_xor(dot, off, 32);
        us  += __shfl_xor(us,  off, 32);
        is  += __shfl_xor(is,  off, 32);
    }
    if (lane == 0) {
        float pred = dot + ubias[u] + ibias[it] + avg[0];
        float diff = pred - ratings[b];
        atomicAdd(acc + 0, diff * diff);
        atomicAdd(acc + 1, us);
        atomicAdd(acc + 2, is);
    }
}

__global__ void finalize_kernel(const float* __restrict__ acc, float* __restrict__ out) {
    if (blockIdx.x == 0 && threadIdx.x == 0) {
        float loss2 = acc[0] / (float)B_CNT;
        float l2 = LAMADA * (acc[1] / ((float)B_CNT * DD) + acc[2] / ((float)B_CNT * DD));
        out[0] = loss2 + l2;   // loss
        out[1] = loss2;        // loss2
    }
}

// ---------------------------------------------------------------------------
extern "C" void kernel_launch(void* const* d_in, const int* in_sizes, int n_in,
                              void* d_out, int out_size, void* d_ws, size_t ws_size,
                              hipStream_t stream) {
    const float* ue      = (const float*)d_in[0];
    const float* ie      = (const float*)d_in[1];
    const float* di      = (const float*)d_in[2];
    const float* dj      = (const float*)d_in[3];
    const int*   rows    = (const int*)d_in[4];
    const int*   cols    = (const int*)d_in[5];
    const float* vals    = (const float*)d_in[6];
    const float* W1      = (const float*)d_in[7];
    const float* b1      = (const float*)d_in[8];
    const float* W2      = (const float*)d_in[9];
    const float* b2      = (const float*)d_in[10];
    const float* ubias   = (const float*)d_in[11];
    const float* ibias   = (const float*)d_in[12];
    const float* avg     = (const float*)d_in[13];
    const int*   user0   = (const int*)d_in[14];
    const int*   item0   = (const int*)d_in[15];
    const float* ratings = (const float*)d_in[16];

    const size_t UD = (size_t)U_CNT * DD;   // 12.8M floats
    const size_t ID = (size_t)I_CNT * DD;   //  6.4M floats
    float* ws  = (float*)d_ws;              // needs ~282 MB
    float* s_u = ws;             ws += UD;  // user SpMM accumulator (reused)
    float* g1u = ws;             ws += UD;
    float* gu  = ws;             ws += UD;
    float* guf = ws;             ws += UD;
    float* s_i = ws;             ws += ID;  // item SpMM accumulator (reused)
    float* g1i = ws;             ws += ID;
    float* gi  = ws;             ws += ID;
    float* acc = ws;             ws += 4;

    const int T = 256;
    dim3 blk(T);
    auto blocks = [](size_t n, int t) { return dim3((unsigned)((n + t - 1) / t)); };
    const dim3 spmmGrid((unsigned)(((size_t)NNZ_CNT * 32 + T - 1) / T));

    // --- hop 1 ---
    zero_kernel<<<blocks(UD, T), blk, 0, stream>>>(s_u, UD);
    zero_kernel<<<blocks(ID, T), blk, 0, stream>>>(s_i, ID);
    spmm_kernel<<<spmmGrid, blk, 0, stream>>>(s_u, ie, cols, rows, vals, NNZ_CNT); // A   @ ie
    spmm_kernel<<<spmmGrid, blk, 0, stream>>>(s_i, ue, rows, cols, vals, NNZ_CNT); // A^T @ ue
    ew_hop1_kernel<<<blocks(UD, T), blk, 0, stream>>>(g1u, s_u, ue, di, UD);
    ew_hop1_kernel<<<blocks(ID, T), blk, 0, stream>>>(g1i, s_i, ie, dj, ID);

    // --- hop 2 (reuse accumulators) ---
    zero_kernel<<<blocks(UD, T), blk, 0, stream>>>(s_u, UD);
    zero_kernel<<<blocks(ID, T), blk, 0, stream>>>(s_i, ID);
    spmm_kernel<<<spmmGrid, blk, 0, stream>>>(s_u, g1i, cols, rows, vals, NNZ_CNT);
    spmm_kernel<<<spmmGrid, blk, 0, stream>>>(s_i, g1u, rows, cols, vals, NNZ_CNT);
    ew_hop2_kernel<<<blocks(UD, T), blk, 0, stream>>>(gu, ue, g1u, s_u, di, UD);
    ew_hop2_kernel<<<blocks(ID, T), blk, 0, stream>>>(gi, ie, g1i, s_i, dj, ID);

    // --- WMMA MLP filter: 64 rows per 128-thread block ---
    mlp_wmma_kernel<<<dim3((U_CNT + 63) / 64), dim3(128), 0, stream>>>(
        gu, W1, b1, W2, b2, guf, U_CNT);

    // --- scoring + loss ---
    zero_kernel<<<dim3(1), blk, 0, stream>>>(acc, 4);
    predict_kernel<<<dim3((B_CNT * 32 + T - 1) / T), blk, 0, stream>>>(
        guf, gi, user0, item0, ubias, ibias, avg, ratings, acc, B_CNT);
    finalize_kernel<<<dim3(1), dim3(1), 0, stream>>>(acc, (float*)d_out);
}
